// AdHocGraphQ_88837103551519
// MI455X (gfx1250) — matrix operand order repaired and checked by reading, not compiled, not run on hardware
//
#include <hip/hip_runtime.h>

typedef __attribute__((ext_vector_type(16))) __bf16 v16bf;
typedef __attribute__((ext_vector_type(8)))  __bf16 v8bf;
typedef __attribute__((ext_vector_type(4)))  __bf16 v4bf;
typedef __attribute__((ext_vector_type(8)))  float  v8f;
typedef __attribute__((ext_vector_type(4)))  float  v4f;

constexpr int N_AGENT  = 32768;
constexpr int MAX_NBRS = 16;
constexpr int F_NBR = 32, F_AGENT = 32, F_HOP = 8;
constexpr int H     = 128;
constexpr int N_POW = 4;
constexpr int QCOLS = MAX_NBRS * N_POW + 1;   // 65

// ---- transposed/padded bf16 weight cache in workspace ----------------------
// each is [Npad][Kpad], N-major, zero-padded
constexpr int WT_MSG = 0;                       // [128][64]
constexpr int WT_UPD = WT_MSG + 128 * 64;       // [128][160]
constexpr int WT_X   = WT_UPD + 128 * 160;      // [384][128]
constexpr int WT_H   = WT_X   + 384 * 128;      // [384][128]
constexpr int WT_E1  = WT_H   + 384 * 128;      // [128][192]
constexpr int WT_E2  = WT_E1  + 128 * 192;      // [16][128]

// ---- WMMA fragment helpers (CDNA5 16x16x32 bf16, wave32) -------------------
// A (16xK tile staged in LDS as bf16, row-major, ldk % 16 == 0):
//   lane 0-15 : row M=lane,    K = k0 .. k0+15  (contiguous)
//   lane16-31 : row M=lane-16, K = k0+16 .. k0+31
__device__ inline v16bf lds_a_frag(const __bf16* a16, int ldk, int k0) {
  const int lane = threadIdx.x & 31;
  const int row  = lane & 15;
  const int kb   = k0 + ((lane >> 4) << 4);
  const __bf16* p = a16 + row * ldk + kb;
  const v8bf lo = *(const v8bf*)(p);
  const v8bf hi = *(const v8bf*)(p + 8);
  return __builtin_shufflevector(lo, hi, 0, 1, 2, 3, 4, 5, 6, 7,
                                 8, 9, 10, 11, 12, 13, 14, 15);
}

// B from transposed weight cache Wt[N][Kpad] (bf16, global L2 or LDS copy):
//   lane 0-15 : col N=n0+lane,    K = k0 .. k0+15  (contiguous)
//   lane16-31 : col N=n0+lane-16, K = k0+16 .. k0+31
__device__ inline v16bf bt_frag(const __bf16* __restrict__ Wt, int ldk,
                                int k0, int n0) {
  const int lane = threadIdx.x & 31;
  const int n    = n0 + (lane & 15);
  const int kb   = k0 + ((lane >> 4) << 4);
  return *(const v16bf*)(Wt + n * ldk + kb);
}

__device__ inline v8f wmma_bf16(v16bf a, v16bf b, v8f c) {
  return __builtin_amdgcn_wmma_f32_16x16x32_bf16(false, a, false, b,
                                                 (short)0, c, false, false);
}

__device__ inline float fsigmoid(float x) { return 1.f / (1.f + __expf(-x)); }

// float4 -> 4x bf16 staging (global_load_b128 + 2x v_cvt_pk + ds_store_b64)
__device__ inline void stage4(__bf16* dst, const float* __restrict__ src) {
  const v4f v = *(const v4f*)(src);
  v4bf b;
  b[0] = (__bf16)v[0]; b[1] = (__bf16)v[1];
  b[2] = (__bf16)v[2]; b[3] = (__bf16)v[3];
  *(v4bf*)(dst) = b;
}
__device__ inline void zero4(__bf16* dst) {
  *(v4bf*)(dst) = (v4bf)(__bf16)0.f;
}
// 16B LDS <- global copy for bf16 weight panels
__device__ inline void copy16(void* dst, const void* src) {
  *(uint4*)dst = *(const uint4*)src;
}

// ---- Kernel 0: transpose+pad all weights to bf16 [N][Kpad] in workspace ----
__device__ inline void tpose_one(const float* __restrict__ W, __bf16* __restrict__ Wt,
                                 int K, int N, int Kpad, int Npad, int srcld,
                                 int tid, int nth) {
  for (int idx = tid; idx < Npad * Kpad; idx += nth) {
    const int n = idx / Kpad, k = idx - n * Kpad;
    const float v = (k < K && n < N) ? W[k * srcld + n] : 0.f;
    Wt[idx] = (__bf16)v;
  }
}

__global__ __launch_bounds__(256) void weight_prep_kernel(
    const float* __restrict__ W_msg, const float* __restrict__ W_upd,
    const float* __restrict__ W_x,   const float* __restrict__ W_h,
    const float* __restrict__ W_e1,  const float* __restrict__ W_e2,
    __bf16* __restrict__ wt) {
  const int tid = blockIdx.x * blockDim.x + threadIdx.x;
  const int nth = gridDim.x * blockDim.x;
  tpose_one(W_msg, wt + WT_MSG, F_NBR + F_HOP, H,     64,  H,     H,     tid, nth);
  tpose_one(W_upd, wt + WT_UPD, H + F_AGENT,   H,     160, H,     H,     tid, nth);
  tpose_one(W_x,   wt + WT_X,   H,             3 * H, 128, 3 * H, 3 * H, tid, nth);
  tpose_one(W_h,   wt + WT_H,   H,             3 * H, 128, 3 * H, 3 * H, tid, nth);
  tpose_one(W_e1,  wt + WT_E1,  F_NBR + H + F_HOP, H, 192, H,     H,     tid, nth);
  tpose_one(W_e2,  wt + WT_E2,  H,             N_POW, 128, 16,    N_POW, tid, nth);
}

// ---- Kernel 1: edge message MLP + segment-sum (one agent == one M-tile) ----
__global__ __launch_bounds__(128) void msg_agg_kernel(
    const float* __restrict__ feat_nbr, const float* __restrict__ edge_feat,
    const __bf16* __restrict__ wt, const float* __restrict__ b_msg,
    const int* __restrict__ src_idx, float* __restrict__ agg) {
  constexpr int KP = 64;                       // pad K: 40 -> 64
  __shared__ alignas(32) __bf16 s_a[4][16 * KP];
  __shared__ alignas(32) __bf16 s_w[128 * KP]; // wt_msg copy, shared by block
  const int wave  = threadIdx.x >> 5;
  const int lane  = threadIdx.x & 31;
  const int agent = blockIdx.x * 4 + wave;
  __bf16* as = s_a[wave];

  // block-cooperative weight panel copy (16KB)
  for (int i = threadIdx.x; i < 128 * KP / 8; i += 128)
    copy16(s_w + i * 8, wt + WT_MSG + i * 8);

  // stage A = [feat_nbr[src] | edge_feat | 0] as bf16, float4-vectorized
  for (int idx = lane; idx < 16 * 8; idx += 32) {          // 32 cols = 8 chunks
    const int row = idx >> 3, c4 = idx & 7;
    const int s = src_idx[agent * MAX_NBRS + row];
    stage4(as + row * KP + c4 * 4, feat_nbr + s * F_NBR + c4 * 4);
  }
  {                                                         // 8 cols = 2 chunks
    const int row = lane >> 1, c4 = lane & 1;
    stage4(as + row * KP + F_NBR + c4 * 4,
           edge_feat + (agent * MAX_NBRS + row) * F_HOP + c4 * 4);
  }
  for (int idx = lane; idx < 16 * 6; idx += 32) {          // pad 40..63
    const int row = idx / 6, c4 = idx - row * 6;
    zero4(as + row * KP + 40 + c4 * 4);
  }
  __syncthreads();

  const v16bf a0 = lds_a_frag(as, KP, 0);
  const v16bf a1 = lds_a_frag(as, KP, 32);
  const int cn = lane & 15;

#pragma unroll
  for (int t = 0; t < 8; ++t) {
    v8f c = {};
    c = wmma_bf16(a0, bt_frag(s_w, KP, 0,  t * 16), c);
    c = wmma_bf16(a1, bt_frag(s_w, KP, 32, t * 16), c);
    const float bias = b_msg[t * 16 + cn];
    float s = 0.f;
#pragma unroll
    for (int r = 0; r < 8; ++r) s += fmaxf(c[r] + bias, 0.f);
    s += __shfl_xor(s, 16, 32);                 // fold M 0-7 with M 8-15
    if (lane < 16) agg[agent * H + t * 16 + lane] = s;
  }
}

// ---- Kernel 2: node update MLP + GRU cell (16 agents per wave) -------------
__global__ __launch_bounds__(128) void update_gru_kernel(
    const float* __restrict__ agg, const float* __restrict__ feat_agent,
    const float* __restrict__ h, const __bf16* __restrict__ wt,
    const float* __restrict__ b_upd,
    const float* __restrict__ b_ih, const float* __restrict__ b_hh,
    float* __restrict__ h_new) {
  constexpr int K1 = H + F_AGENT;              // 160 (exact 5 k-steps)
  __shared__ alignas(32) __bf16 s_a1[4][16 * K1];
  __shared__ alignas(32) __bf16 s_x [4][16 * H];
  __shared__ alignas(32) __bf16 s_h [4][16 * H];
  const int wave = threadIdx.x >> 5;
  const int lane = threadIdx.x & 31;
  const int a0   = (blockIdx.x * 4 + wave) * 16;
  __bf16* a1 = s_a1[wave];
  __bf16* xs = s_x[wave];
  __bf16* hs = s_h[wave];
  const __bf16* wt_upd = wt + WT_UPD;
  const __bf16* wt_x   = wt + WT_X;
  const __bf16* wt_h   = wt + WT_H;

  for (int idx = lane; idx < 16 * 32; idx += 32) {         // H = 32 chunks
    const int row = idx >> 5, c4 = idx & 31;
    stage4(a1 + row * K1 + c4 * 4, agg + (a0 + row) * H + c4 * 4);
    stage4(hs + row * H  + c4 * 4, h   + (a0 + row) * H + c4 * 4);
  }
  for (int idx = lane; idx < 16 * 8; idx += 32) {          // F_AGENT = 8 chunks
    const int row = idx >> 3, c4 = idx & 7;
    stage4(a1 + row * K1 + H + c4 * 4,
           feat_agent + (a0 + row) * F_AGENT + c4 * 4);
  }
  __syncthreads();

  const int cn = lane & 15;
  const int hi = (lane >> 4) << 3;

  // x = relu([agg|feat_agent] @ W_upd + b_upd), restaged as bf16 A-tile
#pragma unroll
  for (int t = 0; t < 8; ++t) {
    v8f c = {};
#pragma unroll
    for (int k = 0; k < 5; ++k)
      c = wmma_bf16(lds_a_frag(a1, K1, k * 32),
                    bt_frag(wt_upd, K1, k * 32, t * 16), c);
    const float bias = b_upd[t * 16 + cn];
#pragma unroll
    for (int r = 0; r < 8; ++r)
      xs[(r + hi) * H + t * 16 + cn] = (__bf16)fmaxf(c[r] + bias, 0.f);
  }
  __syncthreads();

  // GRU: per 16-column output tile, compute all six gate GEMM tiles
#pragma unroll 1
  for (int j = 0; j < 8; ++j) {
    v8f ir = {}, iz = {}, in = {}, hr = {}, hz = {}, hn = {};
#pragma unroll
    for (int k = 0; k < 4; ++k) {
      const v16bf ax = lds_a_frag(xs, H, k * 32);
      const v16bf ah = lds_a_frag(hs, H, k * 32);
      ir = wmma_bf16(ax, bt_frag(wt_x, H, k * 32,         j * 16), ir);
      iz = wmma_bf16(ax, bt_frag(wt_x, H, k * 32, H     + j * 16), iz);
      in = wmma_bf16(ax, bt_frag(wt_x, H, k * 32, 2 * H + j * 16), in);
      hr = wmma_bf16(ah, bt_frag(wt_h, H, k * 32,         j * 16), hr);
      hz = wmma_bf16(ah, bt_frag(wt_h, H, k * 32, H     + j * 16), hz);
      hn = wmma_bf16(ah, bt_frag(wt_h, H, k * 32, 2 * H + j * 16), hn);
    }
    const int col = j * 16 + cn;
    const float bir = b_ih[col],         bhr = b_hh[col];
    const float biz = b_ih[H + col],     bhz = b_hh[H + col];
    const float bin = b_ih[2 * H + col], bhn = b_hh[2 * H + col];
#pragma unroll
    for (int r = 0; r < 8; ++r) {
      const int row = a0 + r + hi;
      const float rg = fsigmoid(ir[r] + bir + hr[r] + bhr);
      const float zg = fsigmoid(iz[r] + biz + hz[r] + bhz);
      const float ng = tanhf(in[r] + bin + rg * (hn[r] + bhn));
      const float hv = h[row * H + col];
      h_new[row * H + col] = (1.f - zg) * ng + zg * hv;
    }
  }
}

// ---- Kernel 3: edge head MLP + q heads (one agent per wave) ----------------
__global__ __launch_bounds__(128) void edge_out_kernel(
    const float* __restrict__ feat_nbr, const float* __restrict__ edge_feat,
    const float* __restrict__ h_new, const __bf16* __restrict__ wt,
    const float* __restrict__ b_e1, const float* __restrict__ b_e2,
    const float* __restrict__ W_a,  const float* __restrict__ b_a,
    const int* __restrict__ src_idx, float* __restrict__ q_out) {
  constexpr int K3 = 192;                      // pad(32 + 128 + 8) -> 6 k-steps
  __shared__ alignas(32) __bf16 s_a[4][16 * K3];
  __shared__ alignas(32) __bf16 s_z[4][16 * H];
  __shared__ alignas(32) __bf16 s_w1[128 * K3]; // wt_e1 copy (48KB, block-shared)
  __shared__ alignas(32) __bf16 s_w2[16 * H];   // wt_e2 copy (4KB)
  const int wave  = threadIdx.x >> 5;
  const int lane  = threadIdx.x & 31;
  const int agent = blockIdx.x * 4 + wave;
  __bf16* as = s_a[wave];
  __bf16* zs = s_z[wave];

  // block-cooperative weight panel copies
  for (int i = threadIdx.x; i < 128 * K3 / 8; i += 128)
    copy16(s_w1 + i * 8, wt + WT_E1 + i * 8);
  for (int i = threadIdx.x; i < 16 * H / 8; i += 128)
    copy16(s_w2 + i * 8, wt + WT_E2 + i * 8);

  // stage A = [feat_nbr[src] | h_new[agent] | edge_feat | 0]
  for (int idx = lane; idx < 16 * 8; idx += 32) {
    const int row = idx >> 3, c4 = idx & 7;
    const int s = src_idx[agent * MAX_NBRS + row];
    stage4(as + row * K3 + c4 * 4, feat_nbr + s * F_NBR + c4 * 4);
  }
  for (int idx = lane; idx < 16 * 32; idx += 32) {
    const int row = idx >> 5, c4 = idx & 31;
    stage4(as + row * K3 + F_NBR + c4 * 4,
           h_new + agent * H + c4 * 4);                    // dst == agent
  }
  {
    const int row = lane >> 1, c4 = lane & 1;
    stage4(as + row * K3 + F_NBR + H + c4 * 4,
           edge_feat + (agent * MAX_NBRS + row) * F_HOP + c4 * 4);
  }
  for (int idx = lane; idx < 16 * 6; idx += 32) {          // pad 168..191
    const int row = idx / 6, c4 = idx - row * 6;
    zero4(as + row * K3 + 168 + c4 * 4);
  }
  __syncthreads();

  const int cn = lane & 15;
  const int hi = (lane >> 4) << 3;

  float aw = 0.f;                               // agg2 . W_a partials (lanes<16)
#pragma unroll 1
  for (int t = 0; t < 8; ++t) {
    v8f c = {};
#pragma unroll
    for (int k = 0; k < 6; ++k)
      c = wmma_bf16(lds_a_frag(as, K3, k * 32),
                    bt_frag(s_w1, K3, k * 32, t * 16), c);
    const float bias = b_e1[t * 16 + cn];
    float s = 0.f;
#pragma unroll
    for (int r = 0; r < 8; ++r) {
      const float v = fmaxf(c[r] + bias, 0.f);
      zs[(r + hi) * H + t * 16 + cn] = (__bf16)v;  // restage ze for q-head GEMM
      s += v;
    }
    s += __shfl_xor(s, 16, 32);
    if (lane < 16) aw += s * W_a[t * 16 + lane];
  }
  __syncthreads();

  // nbr_out = ze @ W_e2 : N padded 4 -> 16 (padded cols are zero)
  v8f q = {};
#pragma unroll
  for (int k = 0; k < 4; ++k)
    q = wmma_bf16(lds_a_frag(zs, H, k * 32),
                  bt_frag(s_w2, H, k * 32, 0), q);
  if (cn < N_POW) {
    const float bias = b_e2[cn];
#pragma unroll
    for (int r = 0; r < 8; ++r) {
      const int e = r + hi;
      q_out[agent * QCOLS + e * N_POW + cn] = q[r] + bias;
    }
  }

  // agent_out = [agg2 | h_new] . W_a + b_a
  float hw = 0.f;
#pragma unroll
  for (int c4 = 0; c4 < 4; ++c4) {
    const int col = c4 * 32 + lane;
    hw += h_new[agent * H + col] * W_a[H + col];
  }
  float tot = (lane < 16 ? aw : 0.f) + hw;
#pragma unroll
  for (int m = 16; m >= 1; m >>= 1) tot += __shfl_xor(tot, m, 32);
  if (lane == 0) q_out[agent * QCOLS + MAX_NBRS * N_POW] = tot + b_a[0];
}

// ---- launch ----------------------------------------------------------------
extern "C" void kernel_launch(void* const* d_in, const int* in_sizes, int n_in,
                              void* d_out, int out_size, void* d_ws, size_t ws_size,
                              hipStream_t stream) {
  const float* feat_nbr   = (const float*)d_in[0];
  const float* feat_agent = (const float*)d_in[1];
  const float* edge_feat  = (const float*)d_in[2];
  const float* h          = (const float*)d_in[3];
  const float* W_msg = (const float*)d_in[4];
  const float* b_msg = (const float*)d_in[5];
  const float* W_upd = (const float*)d_in[6];
  const float* b_upd = (const float*)d_in[7];
  const float* W_x   = (const float*)d_in[8];
  const float* W_h   = (const float*)d_in[9];
  const float* b_ih  = (const float*)d_in[10];
  const float* b_hh  = (const float*)d_in[11];
  const float* W_e1  = (const float*)d_in[12];
  const float* b_e1  = (const float*)d_in[13];
  const float* W_e2  = (const float*)d_in[14];
  const float* b_e2  = (const float*)d_in[15];
  const float* W_a   = (const float*)d_in[16];
  const float* b_a   = (const float*)d_in[17];
  const int*   src_idx = (const int*)d_in[18];

  float*  q_out = (float*)d_out;                               // [N, 65]
  float*  h_new = (float*)d_out + (size_t)N_AGENT * QCOLS;     // [N, 128]
  float*  agg   = (float*)d_ws;                                // [N, 128] scratch
  __bf16* wt    = (__bf16*)((char*)d_ws +
                            (size_t)N_AGENT * H * sizeof(float)); // weight cache

  const dim3 blk(128);                                         // 4 waves / block
  weight_prep_kernel<<<128, 256, 0, stream>>>(
      W_msg, W_upd, W_x, W_h, W_e1, W_e2, wt);
  msg_agg_kernel<<<N_AGENT / 4, blk, 0, stream>>>(
      feat_nbr, edge_feat, wt, b_msg, src_idx, agg);
  update_gru_kernel<<<N_AGENT / 64, blk, 0, stream>>>(
      agg, feat_agent, h, wt, b_upd, b_ih, b_hh, h_new);
  edge_out_kernel<<<N_AGENT / 4, blk, 0, stream>>>(
      feat_nbr, edge_feat, h_new, wt, b_e1, b_e2, W_a, b_a,
      src_idx, q_out);
}
